// MultiheadSelfAttentionV0_74620761800950
// MI455X (gfx1250) — matrix operand order repaired and checked by reading, not compiled, not run on hardware
//
#include <hip/hip_runtime.h>
#include <cstdint>

typedef _Float16 half_t;
typedef __attribute__((ext_vector_type(16))) _Float16 v16h;
typedef __attribute__((ext_vector_type(8)))  _Float16 v8h;
typedef __attribute__((ext_vector_type(8)))  float    v8f;

typedef int          v4i   __attribute__((ext_vector_type(4)));
typedef unsigned int u32x4 __attribute__((ext_vector_type(4)));
typedef int          i32x4 __attribute__((ext_vector_type(4)));
typedef int          i32x8 __attribute__((ext_vector_type(8)));

#define D_MODEL 768
#define S_LEN   4096
#define BATCH   2
#define HEADS   12
#define HDIM    64
#define MROWS   (BATCH * S_LEN)   // 8192

#define GLOBAL_AS __attribute__((address_space(1)))
#define LDS_AS    __attribute__((address_space(3)))

// CDNA5 async/TDM paths, guarded so an absent builtin falls back to the
// proven synchronous staging instead of breaking the build.
#if __has_builtin(__builtin_amdgcn_global_load_async_to_lds_b128)
#define HAVE_ASYNC_LDS 1
#else
#define HAVE_ASYNC_LDS 0
#endif
#if __has_builtin(__builtin_amdgcn_tensor_load_to_lds)
#define HAVE_TDM 1
#else
#define HAVE_TDM 0
#endif

__device__ __forceinline__ v8f wmma_f16(v16h a, v16h b, v8f c) {
  // D = A(16x32 f16) * B(32x16 f16) + C(16x16 f32)
  return __builtin_amdgcn_wmma_f32_16x16x32_f16(false, a, false, b, (short)0, c,
                                                false, false);
}

__device__ __forceinline__ v16h cat8(v8h lo, v8h hi) {
  v16h r;
#pragma unroll
  for (int i = 0; i < 8; ++i) { r[i] = lo[i]; r[i + 8] = hi[i]; }
  return r;
}

// reductions across the 16 lanes of one lane-half (N direction of a C tile)
__device__ __forceinline__ float rowmax16(float v) {
#pragma unroll
  for (int m = 1; m <= 8; m <<= 1) v = fmaxf(v, __shfl_xor(v, m, 32));
  return v;
}
__device__ __forceinline__ float rowsum16(float v) {
#pragma unroll
  for (int m = 1; m <= 8; m <<= 1) v += __shfl_xor(v, m, 32);
  return v;
}

__device__ __forceinline__ void wait_async_lds() {
#if __has_builtin(__builtin_amdgcn_s_wait_asynccnt)
  __builtin_amdgcn_s_wait_asynccnt(0);
#else
  asm volatile("s_wait_asynccnt 0x0" ::: "memory");
#endif
}

#if HAVE_ASYNC_LDS
__device__ __forceinline__ void async_cp16(const half_t* g, half_t* l) {
  __builtin_amdgcn_global_load_async_to_lds_b128(
      (GLOBAL_AS v4i*)(void*)g, (LDS_AS v4i*)(void*)l, 0, 0);
}
#endif

// ---------------------------------------------------------------- convert
__global__ void cvt_f32_f16(const float* __restrict__ in, half_t* __restrict__ out,
                            int n) {
  int i = blockIdx.x * blockDim.x + threadIdx.x;
  int stride = gridDim.x * blockDim.x;
  for (; i < n; i += stride) out[i] = (half_t)in[i];
}

// ---------------------------------------------------------------- GEMM
// Out[M,N] = A[M,768] * W[N,768]^T ; M = 8192, N = 768.
// MODE 0: f16 row-major | MODE 1: f16 per-head V^T [B][H][dv][S] | MODE 2: f32 row-major
template <int MODE>
__global__ __launch_bounds__(256) void gemm_xwT(const half_t* __restrict__ A,
                                                const half_t* __restrict__ W,
                                                void* __restrict__ out) {
  __shared__ __align__(64) half_t As[128 * 32];
  __shared__ __align__(64) half_t Bs[64 * 32];
  const int tid  = threadIdx.x;
  const int wid  = tid >> 5, lane = tid & 31;
  const int wm   = wid >> 1, wn = wid & 1;
  const int sel  = lane >> 4, ln = lane & 15;
  const int m0   = blockIdx.y * 128, n0 = blockIdx.x * 64;

  v8f acc[2][2] = {{{}, {}}, {{}, {}}};

  for (int k0 = 0; k0 < D_MODEL; k0 += 32) {
    // ---- stage A tile 128x32 ------------------------------------------
#if HAVE_ASYNC_LDS
    {
      const half_t* gp = A + (size_t)(m0 + (tid >> 1)) * D_MODEL + k0 + (tid & 1) * 16;
      half_t* lp = &As[(tid >> 1) * 32 + (tid & 1) * 16];
      async_cp16(gp, lp);
      async_cp16(gp + 8, lp + 8);
    }
#else
    *(v16h*)&As[(tid >> 1) * 32 + (tid & 1) * 16] =
        *(const v16h*)(A + (size_t)(m0 + (tid >> 1)) * D_MODEL + k0 + (tid & 1) * 16);
#endif

    // ---- stage B tile 64x32 -------------------------------------------
#if HAVE_TDM
    if (tid < 32) {
      // Tensor Data Mover: 2D tile (tile_dim0=32 elems of 2B, tile_dim1=64 rows)
      // out of the 768x768 f16 weight matrix, row stride 768 elems.
      const size_t gaddr =
          (size_t)(const void*)(W + (size_t)n0 * D_MODEL + (size_t)k0);
      const unsigned lds_base = (unsigned)(size_t)(LDS_AS half_t*)&Bs[0];
      u32x4 g0;
      g0[0] = 1u;  // count=1, is_restore=0, gather off
      g0[1] = lds_base;
      g0[2] = (unsigned)gaddr;
      g0[3] = (unsigned)((gaddr >> 32) & 0x1FFFFFFu) | (2u << 30);  // type=2
      i32x8 g1;
      g1[0] = (int)(1u << 16);        // workgroup_mask=0, data_size=1 (2B)
      g1[1] = (int)(768u << 16);      // tensor_dim0[15:0]=768
      g1[2] = (int)(768u << 16);      // tensor_dim0 hi=0 | tensor_dim1[15:0]=768
      g1[3] = (int)(32u << 16);       // tensor_dim1 hi=0 | tile_dim0=32
      g1[4] = 64;                     // tile_dim1=64, tile_dim2=0
      g1[5] = 768;                    // tensor_dim0_stride=768
      g1[6] = 0;
      g1[7] = 0;
      i32x4 g2 = {0, 0, 0, 0}, g3 = {0, 0, 0, 0};
#if __clang_major__ <= 22
      __builtin_amdgcn_tensor_load_to_lds(g0, g1, g2, g3, 0);
#else
      i32x8 g4 = {0, 0, 0, 0, 0, 0, 0, 0};
      __builtin_amdgcn_tensor_load_to_lds(g0, g1, g2, g3, g4, 0);
#endif
      __builtin_amdgcn_s_wait_tensorcnt((short)0);
    }
#elif HAVE_ASYNC_LDS
    {
      const half_t* gp = W + (size_t)(n0 + (tid >> 2)) * D_MODEL + k0 + (tid & 3) * 8;
      async_cp16(gp, &Bs[(tid >> 2) * 32 + (tid & 3) * 8]);
    }
#else
    *(v8h*)&Bs[(tid >> 2) * 32 + (tid & 3) * 8] =
        *(const v8h*)(W + (size_t)(n0 + (tid >> 2)) * D_MODEL + k0 + (tid & 3) * 8);
#endif

#if HAVE_ASYNC_LDS
    wait_async_lds();  // our async writes to LDS complete before the barrier
#endif
    __syncthreads();

    // ---- fragments + WMMA ---------------------------------------------
    v16h af[2], bf[2];
#pragma unroll
    for (int mt = 0; mt < 2; ++mt) {
      const int row = wm * 32 + mt * 16 + ln;
      af[mt] = cat8(*(const v8h*)&As[row * 32 + sel * 8],
                    *(const v8h*)&As[row * 32 + 16 + sel * 8]);
    }
#pragma unroll
    for (int nt = 0; nt < 2; ++nt) {
      const int col = wn * 32 + nt * 16 + ln;
      bf[nt] = cat8(*(const v8h*)&Bs[col * 32 + sel * 8],
                    *(const v8h*)&Bs[col * 32 + 16 + sel * 8]);
    }
#pragma unroll
    for (int mt = 0; mt < 2; ++mt)
#pragma unroll
      for (int nt = 0; nt < 2; ++nt)
        acc[mt][nt] = wmma_f16(af[mt], bf[nt], acc[mt][nt]);
    __syncthreads();
  }

#pragma unroll
  for (int mt = 0; mt < 2; ++mt)
#pragma unroll
    for (int nt = 0; nt < 2; ++nt) {
      const int gn = n0 + wn * 32 + nt * 16 + ln;
#pragma unroll
      for (int r = 0; r < 8; ++r) {
        const int gm = m0 + wm * 32 + mt * 16 + r + 8 * sel;
        const float v = acc[mt][nt][r];
        if constexpr (MODE == 0) {
          ((half_t*)out)[(size_t)gm * D_MODEL + gn] = (half_t)v;
        } else if constexpr (MODE == 1) {
          const int b = gm >> 12, s = gm & (S_LEN - 1);
          const int hh = gn >> 6, dv = gn & (HDIM - 1);
          ((half_t*)out)[(((size_t)(b * HEADS + hh)) * HDIM + dv) * S_LEN + s] =
              (half_t)v;
        } else {
          ((float*)out)[(size_t)gm * D_MODEL + gn] = v;
        }
      }
    }
}

// ---------------------------------------------------------------- flash attention
// Q,K: f16 [B,S,768] row-major ; Vt: f16 [B][H][dv=64][S=4096] ; Ctx: f16 [B,S,768]
__global__ __launch_bounds__(128) void flash_attn(const half_t* __restrict__ Q,
                                                  const half_t* __restrict__ K,
                                                  const half_t* __restrict__ Vt,
                                                  half_t* __restrict__ Ctx) {
  __shared__ __align__(64) half_t Plds[4 * 16 * 32];  // per-wave 16x32 P bounce
  const int lane = threadIdx.x & 31;
  const int wid  = threadIdx.x >> 5;
  const int sel  = lane >> 4, ln = lane & 15;
  const int h    = blockIdx.y, bb = blockIdx.z;
  const int q0   = blockIdx.x * 64 + wid * 16;

  // Q fragments (wave's 16 rows, d split 0..31 / 32..63), pre-scaled by 1/sqrt(64)
  const half_t* qrow = Q + ((size_t)(bb * S_LEN + q0 + ln)) * D_MODEL + h * HDIM;
  v16h qa0 = cat8(*(const v8h*)(qrow + sel * 8), *(const v8h*)(qrow + 16 + sel * 8));
  v16h qa1 = cat8(*(const v8h*)(qrow + 32 + sel * 8), *(const v8h*)(qrow + 48 + sel * 8));
#pragma unroll
  for (int i = 0; i < 16; ++i) {
    qa0[i] = qa0[i] * (_Float16)0.125f;   // exact power-of-two scale
    qa1[i] = qa1[i] * (_Float16)0.125f;
  }

  v8f oacc[4] = {{}, {}, {}, {}};
  float Lr[8], Mr[8];
#pragma unroll
  for (int r = 0; r < 8; ++r) { Lr[r] = 0.f; Mr[r] = -1e30f; }

  half_t* Pw = &Plds[wid * 512];
  const int    qi    = q0 + 8 * sel;  // acc element r belongs to query row qi+r
  const size_t kvrow = (size_t)bb * S_LEN;
  const size_t vtrow = ((size_t)(bb * HEADS + h)) * HDIM;

  const int kb_end = (q0 + 15) >> 5;  // causal: last 32-key block touching this tile
  for (int kb = 0; kb <= kb_end; ++kb) {
    const int kn = kb * 32;

    if (kb < kb_end)  // pull next K block toward the caches (global_prefetch_b8)
      __builtin_prefetch(K + (kvrow + kn + 32 + ln) * (size_t)D_MODEL + h * HDIM, 0, 3);

    // ---- scores S = (Q/8) @ K^T for 16q x 32k
    v8f s0 = {}, s1 = {};
    {
      const half_t* kr = K + (kvrow + kn + ln) * (size_t)D_MODEL + h * HDIM;
      s0 = wmma_f16(qa0, cat8(*(const v8h*)(kr + sel * 8), *(const v8h*)(kr + 16 + sel * 8)), s0);
      s0 = wmma_f16(qa1, cat8(*(const v8h*)(kr + 32 + sel * 8), *(const v8h*)(kr + 48 + sel * 8)), s0);
    }
    {
      const half_t* kr = K + (kvrow + kn + 16 + ln) * (size_t)D_MODEL + h * HDIM;
      s1 = wmma_f16(qa0, cat8(*(const v8h*)(kr + sel * 8), *(const v8h*)(kr + 16 + sel * 8)), s1);
      s1 = wmma_f16(qa1, cat8(*(const v8h*)(kr + 32 + sel * 8), *(const v8h*)(kr + 48 + sel * 8)), s1);
    }

    // ---- causal mask + online softmax (rows live across 16 lanes)
    float p0[8], p1[8], alpha[8];
#pragma unroll
    for (int r = 0; r < 8; ++r) {
      const float v0 = (kn + ln <= qi + r) ? s0[r] : -1e30f;
      const float v1 = (kn + 16 + ln <= qi + r) ? s1[r] : -1e30f;
      const float bm   = rowmax16(fmaxf(v0, v1));
      const float mnew = fmaxf(Mr[r], bm);
      alpha[r] = __expf(Mr[r] - mnew);
      const float e0 = __expf(v0 - mnew);
      const float e1 = __expf(v1 - mnew);
      Lr[r] = Lr[r] * alpha[r] + rowsum16(e0 + e1);
      Mr[r] = mnew;
      p0[r] = e0;
      p1[r] = e1;
    }
#pragma unroll
    for (int j = 0; j < 4; ++j)
#pragma unroll
      for (int r = 0; r < 8; ++r) oacc[j][r] *= alpha[r];

    // ---- redistribute P: C-layout -> A-layout via per-wave LDS (wave-private,
    // DS pipe is in-order per wave; explicit counter wait before readback)
#pragma unroll
    for (int r = 0; r < 8; ++r) {
      Pw[(r + 8 * sel) * 32 + ln]      = (half_t)p0[r];
      Pw[(r + 8 * sel) * 32 + 16 + ln] = (half_t)p1[r];
    }
    asm volatile("s_wait_dscnt 0x0" ::: "memory");
    const v16h pa = cat8(*(const v8h*)&Pw[ln * 32 + sel * 8],
                         *(const v8h*)&Pw[ln * 32 + 16 + sel * 8]);

    // ---- O += P @ V  (Vt rows are contiguous in key index)
#pragma unroll
    for (int j = 0; j < 4; ++j) {
      const half_t* vr = Vt + (vtrow + j * 16 + ln) * (size_t)S_LEN + kn + sel * 8;
      oacc[j] = wmma_f16(pa, cat8(*(const v8h*)vr, *(const v8h*)(vr + 16)), oacc[j]);
    }
  }

  // ---- normalize and write context (f16, row-major for the O projection)
#pragma unroll
  for (int r = 0; r < 8; ++r) {
    const float inv = 1.0f / Lr[r];
    half_t* crow = Ctx + ((size_t)(bb * S_LEN + qi + r)) * D_MODEL + h * HDIM;
#pragma unroll
    for (int j = 0; j < 4; ++j) crow[j * 16 + ln] = (half_t)(oacc[j][r] * inv);
  }
}

// ---------------------------------------------------------------- launch
extern "C" void kernel_launch(void* const* d_in, const int* in_sizes, int n_in,
                              void* d_out, int out_size, void* d_ws, size_t ws_size,
                              hipStream_t stream) {
  const float* x  = (const float*)d_in[0];
  const float* wq = (const float*)d_in[1];
  const float* wk = (const float*)d_in[2];
  const float* wv = (const float*)d_in[3];
  const float* wo = (const float*)d_in[4];

  const size_t NX = (size_t)MROWS * D_MODEL;    // 6,291,456
  const size_t NW = (size_t)D_MODEL * D_MODEL;  // 589,824

  half_t* xh  = (half_t*)d_ws;
  half_t* wqh = xh + NX;
  half_t* wkh = wqh + NW;
  half_t* wvh = wkh + NW;
  half_t* woh = wvh + NW;
  half_t* Qh  = woh + NW;
  half_t* Kh  = Qh + NX;
  half_t* Vt  = Kh + NX;   // per-head transposed V
  half_t* Ctx = Vt + NX;   // ~67.6 MB total workspace

  cvt_f32_f16<<<4096, 256, 0, stream>>>(x, xh, (int)NX);
  cvt_f32_f16<<<512, 256, 0, stream>>>(wq, wqh, (int)NW);
  cvt_f32_f16<<<512, 256, 0, stream>>>(wk, wkh, (int)NW);
  cvt_f32_f16<<<512, 256, 0, stream>>>(wv, wvh, (int)NW);
  cvt_f32_f16<<<512, 256, 0, stream>>>(wo, woh, (int)NW);

  const dim3 ggrid(D_MODEL / 64, MROWS / 128);  // (12, 64)
  gemm_xwT<0><<<ggrid, 256, 0, stream>>>(xh, wqh, Qh);
  gemm_xwT<0><<<ggrid, 256, 0, stream>>>(xh, wkh, Kh);
  gemm_xwT<1><<<ggrid, 256, 0, stream>>>(xh, wvh, Vt);

  flash_attn<<<dim3(S_LEN / 64, HEADS, BATCH), 128, 0, stream>>>(Qh, Kh, Vt, Ctx);

  gemm_xwT<2><<<ggrid, 256, 0, stream>>>(Ctx, woh, d_out);
}
